// MoEEncoder_33122787787131
// MI455X (gfx1250) — matrix operand order repaired and checked by reading, not compiled, not run on hardware
//
#include <hip/hip_runtime.h>
#include <hip/hip_bf16.h>
#include <stdint.h>

// MoE encoder for MI455X (gfx1250, wave32, WMMA + TDM).
// GEMM stages: v_wmma_f32_16x16x32_f16 (f16 in, f32 accumulate).
// Head-GEMM A tiles staged via tensor_load_to_lds (TDM, 6-arg clang-23 form).

#define N_TOK  16384
#define D_IN   512
#define D_OUT  512
#define HID    256
#define NE     10
#define M_TILE 64
#define MAX_TILES ((2 * N_TOK) / M_TILE)   // worst-case tiles per expert

typedef __attribute__((ext_vector_type(16))) _Float16 v16h;
typedef __attribute__((ext_vector_type(8)))  float    v8f;
typedef __attribute__((ext_vector_type(4)))  unsigned int v4u;
typedef __attribute__((ext_vector_type(8)))  int      v8i;
typedef __attribute__((ext_vector_type(4)))  int      v4i;
typedef _Float16 half_t;

union AFrag { v16h v; uint4 u[2]; };
union H4 { half_t h[4]; uint2 u; };

// set bits [lo, lo+width) of a little-endian dword array (folds at compile time)
__device__ __forceinline__ void dset(uint32_t* d, int lo, int width, uint64_t val) {
    int i = 0;
    while (i < width) {
        int dw = (lo + i) >> 5, sh = (lo + i) & 31;
        int n = 32 - sh; if (n > width - i) n = width - i;
        uint32_t mask = (n == 32) ? 0xFFFFFFFFu : (((1u << n) - 1u) << sh);
        d[dw] = (d[dw] & ~mask) | (((uint32_t)(val >> i) << sh) & mask);
        i += n;
    }
}

// ---------------------------------------------------------------- converts
__global__ void k_f32_to_f16(const float* __restrict__ src,
                             half_t* __restrict__ dst, int n4) {
    int i = blockIdx.x * blockDim.x + threadIdx.x;
    if (i >= n4) return;
    float4 f = ((const float4*)src)[i];
    H4 o;
    o.h[0] = (half_t)f.x; o.h[1] = (half_t)f.y;
    o.h[2] = (half_t)f.z; o.h[3] = (half_t)f.w;
    ((uint2*)dst)[i] = o.u;
}

// ---------------------------------------------------------------- meta zero
__global__ void k_zero_meta(int* meta) {
    if (threadIdx.x < 32) meta[threadIdx.x] = 0;
}

// ---------------------------------------------------------------- gating
// One wave per token. gate_w cached in LDS. lane0 does softmax/top2.
__global__ __launch_bounds__(256) void k_gate(
    const float* __restrict__ x, const float* __restrict__ gate_w,
    const float* __restrict__ gate_b, int* __restrict__ topk_idx,
    float* __restrict__ topk_w, int* __restrict__ counts) {
    __shared__ float gw[D_IN * NE];
    int tid = threadIdx.x;
    for (int i = tid; i < D_IN * NE; i += 256) gw[i] = gate_w[i];
    __syncthreads();

    int wave = tid >> 5, lane = tid & 31;
    int n = blockIdx.x * 8 + wave;
    const float* xr = x + (size_t)n * D_IN;

    float acc[NE];
#pragma unroll
    for (int e = 0; e < NE; e++) acc[e] = 0.f;
    for (int it = 0; it < D_IN / 32; ++it) {
        int d = it * 32 + lane;
        float xv = xr[d];
#pragma unroll
        for (int e = 0; e < NE; e++) acc[e] += xv * gw[d * NE + e];
    }
#pragma unroll
    for (int off = 16; off >= 1; off >>= 1) {
#pragma unroll
        for (int e = 0; e < NE; e++) acc[e] += __shfl_xor(acc[e], off, 32);
    }
    if (lane == 0) {
        float lg[NE], m = -1e30f;
#pragma unroll
        for (int e = 0; e < NE; e++) { lg[e] = acc[e] + gate_b[e]; m = fmaxf(m, lg[e]); }
        float p[NE], s = 0.f;
#pragma unroll
        for (int e = 0; e < NE; e++) { p[e] = __expf(lg[e] - m); s += p[e]; }
        float inv_s = 1.f / s;
#pragma unroll
        for (int e = 0; e < NE; e++) p[e] *= inv_s;
        int i0 = 0;
#pragma unroll
        for (int e = 1; e < NE; e++) if (p[e] > p[i0]) i0 = e;
        int i1 = (i0 == 0) ? 1 : 0;
#pragma unroll
        for (int e = 0; e < NE; e++) if (e != i0 && p[e] > p[i1]) i1 = e;
        float inv = 1.f / (p[i0] + p[i1]);
        topk_idx[n * 2 + 0] = i0;  topk_idx[n * 2 + 1] = i1;
        topk_w[n * 2 + 0] = p[i0] * inv;  topk_w[n * 2 + 1] = p[i1] * inv;
        atomicAdd(&counts[i0], 1);
        atomicAdd(&counts[i1], 1);
    }
}

// ---------------------------------------------------------------- scan (E=10)
__global__ void k_scan(const int* counts, int* off, int* cursor) {
    if (threadIdx.x == 0) {
        int a = 0;
        for (int e = 0; e < NE; e++) { off[e] = a; cursor[e] = a; a += counts[e]; }
        off[NE] = a;
    }
}

// ---------------------------------------------------------------- scatter
__global__ void k_scatter(const int* __restrict__ topk_idx, int* cursor,
                          int* __restrict__ expert_tokens) {
    int n = blockIdx.x * blockDim.x + threadIdx.x;
    if (n >= N_TOK) return;
#pragma unroll
    for (int k = 0; k < 2; k++) {
        int e = topk_idx[n * 2 + k];
        int pos = atomicAdd(&cursor[e], 1);
        expert_tokens[pos] = (n << 1) | k;
    }
}

// ---------------------------------------------------------------- expert MLP
// One workgroup = 64 gathered tokens of one expert: (x@w1+b1).relu @ w2 + b2,
// scaled by the token's renormalized gate weight, scattered to y[(tok<<1)|slot].
__global__ __launch_bounds__(256) void k_expert_mlp(
    const half_t* __restrict__ xh, const half_t* __restrict__ w1h,
    const float* __restrict__ b1, const half_t* __restrict__ w2h,
    const float* __restrict__ b2, const int* __restrict__ expert_off,
    const int* __restrict__ expert_tokens, const float* __restrict__ topk_w,
    float* __restrict__ y) {
    __shared__ half_t xs[M_TILE][D_IN];   // 64 KB
    __shared__ half_t hs[M_TILE][HID];    // 32 KB
    __shared__ int    ent[M_TILE];
    __shared__ float  tws[M_TILE];

    int e = blockIdx.x / MAX_TILES;
    int t = blockIdx.x % MAX_TILES;
    int seg_lo = expert_off[e], seg_hi = expert_off[e + 1];
    int seg_len = seg_hi - seg_lo;
    int m0 = t * M_TILE;
    if (m0 >= seg_len) return;               // uniform per block -> safe

    int tid = threadIdx.x;
    int lane = tid & 31, wave = tid >> 5;

    if (tid < M_TILE) {
        int idx = m0 + tid;
        int entv = (idx < seg_len) ? expert_tokens[seg_lo + idx] : -1;
        ent[tid] = entv;
        tws[tid] = (entv >= 0) ? topk_w[entv] : 0.f;
    }
    __syncthreads();

    // stage gathered x rows (uint4 = 8 halves; 64 uint4 per row)
    for (int u = tid; u < M_TILE * (D_IN / 8); u += 256) {
        int r = u >> 6, c = u & 63;
        uint4 val;
        int entv = ent[r];
        if (entv >= 0) val = ((const uint4*)(xh + (size_t)(entv >> 1) * D_IN))[c];
        else           val = make_uint4(0u, 0u, 0u, 0u);
        ((uint4*)&xs[r][0])[c] = val;
    }
    __syncthreads();

    const int hi = lane >> 4, ln = lane & 15, kbase = hi * 8;

    // ---- layer 1: [64 x 512] @ [512 x 256] + b1, ReLU -> hs (f16)
    const half_t* w1e = w1h + (size_t)e * D_IN * HID;
    for (int tile = wave; tile < (M_TILE / 16) * (HID / 16); tile += 8) {
        int rt = tile / (HID / 16), ct = tile % (HID / 16);
        float bias = b1[e * HID + ct * 16 + ln];
        v8f acc;
#pragma unroll
        for (int r = 0; r < 8; r++) acc[r] = bias;
        const half_t* arow = &xs[rt * 16 + ln][0];
#pragma unroll 4
        for (int k0 = 0; k0 < D_IN; k0 += 32) {
            AFrag a, b;
            a.u[0] = *(const uint4*)(arow + k0 + kbase);
            a.u[1] = *(const uint4*)(arow + k0 + 16 + kbase);
            const half_t* bcol = w1e + (size_t)k0 * HID + ct * 16 + ln;
#pragma unroll
            for (int s = 0; s < 16; s++)
                b.v[s] = bcol[(size_t)(kbase + s + (s >= 8 ? 8 : 0)) * HID];
            if (k0 + 64 < D_IN)   // warm L0/L2 for the B rows two k-steps ahead
                __builtin_prefetch(w1e + (size_t)(k0 + 64 + lane) * HID + ct * 16, 0, 0);
            acc = __builtin_amdgcn_wmma_f32_16x16x32_f16(
                false, a.v, false, b.v, (short)0, acc, false, false);
        }
#pragma unroll
        for (int r = 0; r < 8; r++)
            hs[rt * 16 + r + 8 * hi][ct * 16 + ln] = (half_t)fmaxf(acc[r], 0.f);
    }
    __syncthreads();

    // ---- layer 2: [64 x 256] @ [256 x 512] + b2, scale by gate weight
    const half_t* w2e = w2h + (size_t)e * HID * D_OUT;
    for (int tile = wave; tile < (M_TILE / 16) * (D_OUT / 16); tile += 8) {
        int rt = tile / (D_OUT / 16), ct = tile % (D_OUT / 16);
        int ncol = ct * 16 + ln;
        float bias = b2[e * D_OUT + ncol];
        v8f acc;
#pragma unroll
        for (int r = 0; r < 8; r++) acc[r] = bias;
        const half_t* arow = &hs[rt * 16 + ln][0];
#pragma unroll 4
        for (int k0 = 0; k0 < HID; k0 += 32) {
            AFrag a, b;
            a.u[0] = *(const uint4*)(arow + k0 + kbase);
            a.u[1] = *(const uint4*)(arow + k0 + 16 + kbase);
            const half_t* bcol = w2e + (size_t)k0 * D_OUT + ct * 16 + ln;
#pragma unroll
            for (int s = 0; s < 16; s++)
                b.v[s] = bcol[(size_t)(kbase + s + (s >= 8 ? 8 : 0)) * D_OUT];
            acc = __builtin_amdgcn_wmma_f32_16x16x32_f16(
                false, a.v, false, b.v, (short)0, acc, false, false);
        }
#pragma unroll
        for (int r = 0; r < 8; r++) {
            int m = rt * 16 + r + 8 * hi;
            int entv = ent[m];
            if (entv >= 0)
                y[(size_t)entv * D_OUT + ncol] = acc[r] * tws[m];
        }
    }
}

// ---------------------------------------------------------------- combine
__global__ void k_combine(const float* __restrict__ y, half_t* __restrict__ ch) {
    int i = blockIdx.x * blockDim.x + threadIdx.x;   // over N*D_OUT/4
    if (i >= N_TOK * D_OUT / 4) return;
    int n = i / (D_OUT / 4), c = i % (D_OUT / 4);
    float4 a = ((const float4*)(y + (size_t)(2 * n) * D_OUT))[c];
    float4 b = ((const float4*)(y + (size_t)(2 * n + 1) * D_OUT))[c];
    H4 o;
    o.h[0] = (half_t)(a.x + b.x); o.h[1] = (half_t)(a.y + b.y);
    o.h[2] = (half_t)(a.z + b.z); o.h[3] = (half_t)(a.w + b.w);
    ((uint2*)ch)[i] = o.u;
}

// ---------------------------------------------------------------- head GEMM
// [16384 x 512] @ [512 x 512] + bias -> f32 output. A tile (64 rows = one
// contiguous 64KB block of ch) staged into LDS by the Tensor Data Mover.
__global__ __launch_bounds__(256) void k_head_gemm(
    const half_t* __restrict__ ch, const half_t* __restrict__ wh,
    const float* __restrict__ bias, float* __restrict__ out) {
    __shared__ half_t as[M_TILE][D_OUT];   // 64 KB
    int m0 = blockIdx.x * M_TILE;
    int tid = threadIdx.x, lane = tid & 31, wave = tid >> 5;

#if __has_builtin(__builtin_amdgcn_tensor_load_to_lds)
    if (tid < 32) {
        // D# (CDNA5 ISA): 1-D copy, 32768 x 2B.
        uint64_t ga = (uint64_t)(uintptr_t)(ch + (size_t)m0 * D_OUT);
        uint32_t lds = (uint32_t)(uintptr_t)&as[0][0];  // flat addr[31:0] == LDS offset
        union { uint32_t w[4]; v4u v; } g0 = {};
        union { uint32_t w[8]; v8i v; } g1 = {}, g4 = {};
        union { uint32_t w[4]; v4i v; } g2 = {}, g3 = {};
        g0.w[0] = 1u;                                         // count=1
        g0.w[1] = lds;                                        // lds_addr
        g0.w[2] = (uint32_t)ga;                               // global_addr[31:0]
        g0.w[3] = ((uint32_t)(ga >> 32) & 0x01FFFFFFu) | (2u << 30);  // [56:32] | type=2
        g1.w[0] = 1u << 16;                                   // data_size = 2 bytes
        dset(g1.w,  48, 32, (uint64_t)(M_TILE * D_OUT));      // tensor_dim0
        dset(g1.w,  80, 32, 1);                               // tensor_dim1
        dset(g1.w, 112, 16, (uint64_t)(M_TILE * D_OUT));      // tile_dim0 = 32768
        dset(g1.w, 160, 48, (uint64_t)(M_TILE * D_OUT));      // tensor_dim0_stride
        // clang-23 6-arg form: (g0, g1, g2, g3, extra_g, cpol)
        __builtin_amdgcn_tensor_load_to_lds(g0.v, g1.v, g2.v, g3.v, g4.v, 0);
        __builtin_amdgcn_s_wait_tensorcnt(0);
    }
#else
    for (int u = tid; u < M_TILE * (D_OUT / 8); u += 256) {
        int r = u >> 6, c = u & 63;
        ((uint4*)&as[r][0])[c] = ((const uint4*)(ch + (size_t)(m0 + r) * D_OUT))[c];
    }
#endif
    __syncthreads();

    const int hi = lane >> 4, ln = lane & 15, kbase = hi * 8;
    for (int tile = wave; tile < (M_TILE / 16) * (D_OUT / 16); tile += 8) {
        int rt = tile / (D_OUT / 16), ct = tile % (D_OUT / 16);
        int ncol = ct * 16 + ln;
        float bv = bias[ncol];
        v8f acc;
#pragma unroll
        for (int r = 0; r < 8; r++) acc[r] = bv;
        const half_t* arow = &as[rt * 16 + ln][0];
#pragma unroll 4
        for (int k0 = 0; k0 < D_OUT; k0 += 32) {
            AFrag a, b;
            a.u[0] = *(const uint4*)(arow + k0 + kbase);
            a.u[1] = *(const uint4*)(arow + k0 + 16 + kbase);
            const half_t* bcol = wh + (size_t)k0 * D_OUT + ct * 16 + ln;
#pragma unroll
            for (int s = 0; s < 16; s++)
                b.v[s] = bcol[(size_t)(kbase + s + (s >= 8 ? 8 : 0)) * D_OUT];
            acc = __builtin_amdgcn_wmma_f32_16x16x32_f16(
                false, a.v, false, b.v, (short)0, acc, false, false);
        }
#pragma unroll
        for (int r = 0; r < 8; r++)
            out[(size_t)(m0 + rt * 16 + r + 8 * hi) * D_OUT + ncol] = acc[r];
    }
}

// ---------------------------------------------------------------- launch
extern "C" void kernel_launch(void* const* d_in, const int* in_sizes, int n_in,
                              void* d_out, int out_size, void* d_ws, size_t ws_size,
                              hipStream_t stream) {
    const float* x      = (const float*)d_in[0];
    const float* gate_w = (const float*)d_in[1];
    const float* gate_b = (const float*)d_in[2];
    const float* w1     = (const float*)d_in[3];
    const float* b1     = (const float*)d_in[4];
    const float* w2     = (const float*)d_in[5];
    const float* b2     = (const float*)d_in[6];
    const float* cls_w  = (const float*)d_in[7];
    const float* cls_b  = (const float*)d_in[8];
    const float* vec_w  = (const float*)d_in[9];
    const float* vec_b  = (const float*)d_in[10];
    float* out = (float*)d_out;

    // workspace layout (256B aligned)
    char* ws = (char*)d_ws;
    size_t o = 0;
    auto take = [&](size_t bytes) { char* p = ws + o; o += (bytes + 255) & ~(size_t)255; return p; };
    half_t* xh   = (half_t*)take((size_t)N_TOK * D_IN * 2);
    half_t* w1h  = (half_t*)take((size_t)NE * D_IN * HID * 2);
    half_t* w2h  = (half_t*)take((size_t)NE * HID * D_OUT * 2);
    half_t* clsh = (half_t*)take((size_t)D_OUT * D_OUT * 2);
    half_t* vech = (half_t*)take((size_t)D_OUT * D_OUT * 2);
    half_t* ch   = (half_t*)take((size_t)N_TOK * D_OUT * 2);
    int*    topk_idx = (int*)take((size_t)N_TOK * 2 * 4);
    float*  topk_w   = (float*)take((size_t)N_TOK * 2 * 4);
    int*    expert_tokens = (int*)take((size_t)N_TOK * 2 * 4);
    int*    meta = (int*)take(256);        // counts[10] | off[11] | cursor[10]
    float*  y    = (float*)take((size_t)N_TOK * 2 * D_OUT * 4);
    int* counts = meta, *eoff = meta + 10, *cursor = meta + 21;

    // 1) zero routing meta
    k_zero_meta<<<1, 32, 0, stream>>>(meta);

    // 2) f32 -> f16 conversions (vectorized x4)
    auto conv = [&](const float* s, half_t* d, int n) {
        k_f32_to_f16<<<(n / 4 + 255) / 256, 256, 0, stream>>>(s, d, n / 4);
    };
    conv(x,     xh,   N_TOK * D_IN);
    conv(w1,    w1h,  NE * D_IN * HID);
    conv(w2,    w2h,  NE * HID * D_OUT);
    conv(cls_w, clsh, D_OUT * D_OUT);
    conv(vec_w, vech, D_OUT * D_OUT);

    // 3) gate + counts
    k_gate<<<N_TOK / 8, 256, 0, stream>>>(x, gate_w, gate_b, topk_idx, topk_w, counts);

    // 4) scan, 5) scatter
    k_scan<<<1, 32, 0, stream>>>(counts, eoff, cursor);
    k_scatter<<<N_TOK / 256, 256, 0, stream>>>(topk_idx, cursor, expert_tokens);

    // 6) per-expert fused MLP (gathered, weighted scatter to y)
    k_expert_mlp<<<NE * MAX_TILES, 256, 0, stream>>>(
        xh, w1h, b1, w2h, b2, eoff, expert_tokens, topk_w, y);

    // 7) combine top-2 contributions -> f16
    k_combine<<<(N_TOK * D_OUT / 4 + 255) / 256, 256, 0, stream>>>(y, ch);

    // 8) two output heads
    k_head_gemm<<<N_TOK / M_TILE, 256, 0, stream>>>(ch, clsh, cls_b, out);
    k_head_gemm<<<N_TOK / M_TILE, 256, 0, stream>>>(ch, vech, vec_b,
                                                    out + (size_t)N_TOK * D_OUT);
}
